// TriMulOptimized_43576738186058
// MI455X (gfx1250) — compile-verified, matrix operand order checked by reading, not run on hardware
//
#include <hip/hip_runtime.h>

// ---------------------------------------------------------------------------
// TriMul for MI455X (gfx1250, wave32, WMMA bf16 16x16x32, async LDS copies)
// ---------------------------------------------------------------------------
constexpr int N   = 384;
constexpr int NN  = N * N;      // 147456 flattened (i,j) rows
constexpr int DIM = 128;        // = HID

typedef __attribute__((ext_vector_type(16))) __bf16 v16bf;
typedef __attribute__((ext_vector_type(8)))  float  v8f;

// ---- bf16 helpers (storage type = unsigned short) -------------------------
static __device__ __forceinline__ unsigned short f2bf(float f) {
    unsigned int u = __float_as_uint(f);
    u += 0x7FFFu + ((u >> 16) & 1u);          // round-to-nearest-even
    return (unsigned short)(u >> 16);
}
static __device__ __forceinline__ float bf2f(unsigned short s) {
    return __uint_as_float(((unsigned int)s) << 16);
}
static __device__ __forceinline__ float sigmoidf(float x) {
    return 1.0f / (1.0f + __expf(-x));
}

// ---- WMMA fragment loader -------------------------------------------------
// Loads a 16x32 bf16 tile (rows x cols, row-major, row stride `ld` elements)
// into the CDNA5 A/B fragment layout (ISA 7.12.2):
//   lane 0-15 : row = lane,     K = 0..7 (v0-3) and 16..23 (v4-7)
//   lane16-31 : row = lane-16,  K = 8..15 (v0-3) and 24..31 (v4-7)
// The same layout serves B (32x16, K x N) when the source holds the 16 output
// columns row-major over K (our W[n][k] and R_d[j][k] cases).
static __device__ __forceinline__ v16bf
load_frag(const unsigned short* __restrict__ p, int ld, int row0, int col0, int lane) {
    const int r  = row0 + (lane & 15);
    const int kh = (lane >> 4) * 8;
    const unsigned short* rp = p + r * ld + col0;
    union { v16bf v; unsigned int u[8]; } f;
#pragma unroll
    for (int reg = 0; reg < 8; ++reg) {
        const int k = (reg >> 2) * 16 + kh + (reg & 3) * 2;  // bf16 pair base
        f.u[reg] = *(const unsigned int*)(rp + k);
    }
    return f.v;
}

static __device__ __forceinline__ v8f wmma_bf16(v16bf a, v16bf b, v8f c) {
    return __builtin_amdgcn_wmma_f32_16x16x32_bf16(
        /*neg_a=*/false, a, /*neg_b=*/false, b,
        /*c_mod=*/(short)0, c, /*reuse_a=*/false, /*reuse_b=*/false);
}

// CDNA5 async global->LDS copy, 16B per lane (ASYNCcnt-tracked, ISA §10 async)
static __device__ __forceinline__ void
async_copy_b128(unsigned lds_off, unsigned long long gaddr) {
    asm volatile("global_load_async_to_lds_b128 %0, %1, off"
                 :: "v"(lds_off), "v"(gaddr) : "memory");
}
static __device__ __forceinline__ void wait_async0() {
    asm volatile("s_wait_asynccnt 0" ::: "memory");
}
static __device__ __forceinline__ unsigned lds_offset(const void* p) {
    return (unsigned)(size_t)p;   // generic LDS addr: low 32 bits = LDS offset
}

// ---------------------------------------------------------------------------
// Kernel 0: convert the 6 weight matrices (5 gates + Wout) fp32 -> bf16
// wb layout (16384 elems each): Wl, Wr, Wlg, Wrg, Wog, Wout
// ---------------------------------------------------------------------------
__global__ void k0_convert_weights(const float* __restrict__ Wl,
                                   const float* __restrict__ Wr,
                                   const float* __restrict__ Wlg,
                                   const float* __restrict__ Wrg,
                                   const float* __restrict__ Wog,
                                   const float* __restrict__ Wout,
                                   unsigned short* __restrict__ wb) {
    const int idx = blockIdx.x * blockDim.x + threadIdx.x;   // 0..98303
    const float* srcs[6] = { Wl, Wr, Wlg, Wrg, Wog, Wout };
    const int which = idx >> 14;          // /16384
    const int off   = idx & 16383;
    wb[idx] = f2bf(srcs[which][off]);
}

// ---------------------------------------------------------------------------
// Kernel 1: LN1 + 5 gated projections, 16 rows of flattened [NN, DIM] / block
// Outputs:
//   left_t / right_t : bf16 [DIM][NN]  (channel-major for the einsum)
//   og               : bf16 [NN][DIM]
// ---------------------------------------------------------------------------
__global__ void
__launch_bounds__(128)
k1_ln_proj(const float* __restrict__ x, const float* __restrict__ mask,
           const float* __restrict__ ln1w, const float* __restrict__ ln1b,
           const unsigned short* __restrict__ wb,
           unsigned short* __restrict__ left_t,
           unsigned short* __restrict__ right_t,
           unsigned short* __restrict__ og) {
    __shared__ float          xs[16][DIM];     // 8 KB
    __shared__ float          mu[16], rs[16], mk[16];
    __shared__ unsigned short xnb[16][DIM];    // 4 KB bf16 normalized rows

    const int tid  = threadIdx.x;
    const int r0   = blockIdx.x * 16;          // first flattened row
    const size_t gbase = (size_t)r0 * DIM;

    // ---- stage x tile -----------------------------------------------------
    for (int i = tid; i < 16 * DIM; i += 128)
        xs[i >> 7][i & 127] = x[gbase + i];
    if (tid < 16) mk[tid] = mask[r0 + tid];
    __syncthreads();

    // ---- per-row LN stats -------------------------------------------------
    if (tid < 16) {
        float s = 0.f, ss = 0.f;
#pragma unroll 4
        for (int c = 0; c < DIM; ++c) { float v = xs[tid][c]; s += v; ss += v * v; }
        const float m = s * (1.0f / DIM);
        const float var = ss * (1.0f / DIM) - m * m;
        mu[tid] = m;
        rs[tid] = __frsqrt_rn(var + 1e-5f);
    }
    __syncthreads();

    // ---- normalize -> bf16 ------------------------------------------------
    for (int i = tid; i < 16 * DIM; i += 128) {
        const int r = i >> 7, c = i & 127;
        const float v = (xs[r][c] - mu[r]) * rs[r] * ln1w[c] + ln1b[c];
        xnb[r][c] = f2bf(v);
    }
    __syncthreads();

    // ---- WMMA projections (wave id made scalar so branches stay uniform:
    //      WMMA requires EXEC all-1s) ---------------------------------------
    const int wave = __builtin_amdgcn_readfirstlane(tid) >> 5;
    const int lane = tid & 31;

    v16bf afr[4];
#pragma unroll
    for (int kt = 0; kt < 4; ++kt)
        afr[kt] = load_frag(&xnb[0][0], DIM, 0, kt * 32, lane);

    // tasks: pair (0=left,1=right,2=outgate) x 8 n-tiles
    for (int task = wave; task < 24; task += 4) {
        const int pair = task >> 3;
        const int nt   = task & 7;
        const unsigned short* Wlin = wb + (pair == 0 ? 0 : pair == 1 ? 16384 : 65536);
        const unsigned short* Wg   = wb + (pair == 0 ? 32768 : 49152);

        v8f acc = {}, accg = {};
#pragma unroll
        for (int kt = 0; kt < 4; ++kt) {
            v16bf b = load_frag(Wlin, DIM, nt * 16, kt * 32, lane);
            acc = wmma_bf16(afr[kt], b, acc);
            if (pair < 2) {
                v16bf bg = load_frag(Wg, DIM, nt * 16, kt * 32, lane);
                accg = wmma_bf16(afr[kt], bg, accg);
            }
        }

        // C layout: lane&15 = n, VGPR v -> m = v + 8*(lane>>4)
        const int n  = lane & 15;
        const int mh = (lane >> 4) * 8;
        const int d  = nt * 16 + n;
#pragma unroll
        for (int v = 0; v < 8; ++v) {
            const int m = v + mh;
            const size_t g = (size_t)(r0 + m);
            if (pair < 2) {
                const float val = acc[v] * mk[m] * sigmoidf(accg[v]);
                unsigned short* dst = (pair == 0) ? left_t : right_t;
                dst[(size_t)d * NN + g] = f2bf(val);
            } else {
                og[g * DIM + d] = f2bf(sigmoidf(acc[v]));
            }
        }
    }
}

// ---------------------------------------------------------------------------
// Kernel 2: per-channel einsum  out_t[d] = L_d (NxN) * R_d^T (NxN)
// grid = (N/64, N/64, DIM); block = 128 (4 waves, 2x2 wave grid).
// Double-buffered GLOBAL_LOAD_ASYNC_TO_LDS_B128 pipeline: issue next k-tile's
// async copies, run WMMAs from the current buffer, s_wait_asynccnt, barrier.
// ---------------------------------------------------------------------------
__global__ void
__launch_bounds__(128)
k2_einsum(const unsigned short* __restrict__ left_t,
          const unsigned short* __restrict__ right_t,
          float* __restrict__ out_t) {
    // 64 rows x 32 k bf16, row stride 40 elems = 80 B (16B multiple for B128)
    __shared__ __align__(16) unsigned short As[2][64][40];
    __shared__ __align__(16) unsigned short Bs[2][64][40];

    const int tid  = threadIdx.x;
    const int wave = __builtin_amdgcn_readfirstlane(tid) >> 5;
    const int lane = tid & 31;
    const int wi   = (wave >> 1) * 2;   // first i sub-tile of this wave
    const int wj   = (wave & 1) * 2;    // first j sub-tile of this wave

    const int d  = blockIdx.z;
    const int j0 = blockIdx.x * 64;
    const int i0 = blockIdx.y * 64;

    const unsigned short* __restrict__ L = left_t  + (size_t)d * NN;
    const unsigned short* __restrict__ R = right_t + (size_t)d * NN;
    float* __restrict__ O = out_t + (size_t)d * NN;

    // stage one 64x32 bf16 tile per operand: 256 x 16B transfers, 2/thread
    auto stage = [&](int buf, int kk) {
#pragma unroll
        for (int t = 0; t < 2; ++t) {
            const int tr  = tid + t * 128;   // 0..255
            const int r   = tr >> 2;
            const int seg = (tr & 3) * 8;    // 16B segment (8 bf16)
            async_copy_b128(lds_offset(&As[buf][r][seg]),
                            (unsigned long long)(const void*)
                                &L[(size_t)(i0 + r) * N + kk + seg]);
            async_copy_b128(lds_offset(&Bs[buf][r][seg]),
                            (unsigned long long)(const void*)
                                &R[(size_t)(j0 + r) * N + kk + seg]);
        }
    };

    stage(0, 0);
    wait_async0();
    __syncthreads();

    v8f acc[2][2] = {{v8f{}, v8f{}}, {v8f{}, v8f{}}};

    int cur = 0;
    for (int kk = 0; kk < N; kk += 32) {
        if (kk + 32 < N)
            stage(cur ^ 1, kk + 32);         // overlap copy with WMMA below

        const v16bf a0 = load_frag(&As[cur][0][0], 40, (wi + 0) * 16, 0, lane);
        const v16bf a1 = load_frag(&As[cur][0][0], 40, (wi + 1) * 16, 0, lane);
        const v16bf b0 = load_frag(&Bs[cur][0][0], 40, (wj + 0) * 16, 0, lane);
        const v16bf b1 = load_frag(&Bs[cur][0][0], 40, (wj + 1) * 16, 0, lane);
        acc[0][0] = wmma_bf16(a0, b0, acc[0][0]);
        acc[0][1] = wmma_bf16(a0, b1, acc[0][1]);
        acc[1][0] = wmma_bf16(a1, b0, acc[1][0]);
        acc[1][1] = wmma_bf16(a1, b1, acc[1][1]);

        wait_async0();                       // next tile landed in LDS
        __syncthreads();
        cur ^= 1;
    }

    const int n  = lane & 15;
    const int mh = (lane >> 4) * 8;
#pragma unroll
    for (int ii = 0; ii < 2; ++ii)
#pragma unroll
        for (int jj = 0; jj < 2; ++jj)
#pragma unroll
            for (int v = 0; v < 8; ++v) {
                const int i = i0 + (wi + ii) * 16 + v + mh;
                const int j = j0 + (wj + jj) * 16 + n;
                O[(size_t)i * N + j] = acc[ii][jj][v];
            }
}

// ---------------------------------------------------------------------------
// Kernel 3: LN2 + out_gate + final projection (x @ Wout^T), 16 rows / block
// ---------------------------------------------------------------------------
__global__ void
__launch_bounds__(128)
k3_out(const float* __restrict__ out_t, const unsigned short* __restrict__ og,
       const float* __restrict__ ln2w, const float* __restrict__ ln2b,
       const unsigned short* __restrict__ wout_bf,
       float* __restrict__ out) {
    __shared__ float          ts[16][DIM];    // gathered [row][d]
    __shared__ float          mu[16], rs[16];
    __shared__ unsigned short ab[16][DIM];

    const int tid = threadIdx.x;
    const int r0  = blockIdx.x * 16;

    // gather: thread tid owns channel d = tid across the 16 rows
#pragma unroll
    for (int m = 0; m < 16; ++m)
        ts[m][tid] = out_t[(size_t)tid * NN + r0 + m];
    __syncthreads();

    if (tid < 16) {
        float s = 0.f, ss = 0.f;
#pragma unroll 4
        for (int c = 0; c < DIM; ++c) { float v = ts[tid][c]; s += v; ss += v * v; }
        const float m = s * (1.0f / DIM);
        const float var = ss * (1.0f / DIM) - m * m;
        mu[tid] = m;
        rs[tid] = __frsqrt_rn(var + 1e-5f);
    }
    __syncthreads();

    for (int i = tid; i < 16 * DIM; i += 128) {
        const int r = i >> 7, c = i & 127;
        float v = (ts[r][c] - mu[r]) * rs[r] * ln2w[c] + ln2b[c];
        v *= bf2f(og[(size_t)(r0 + r) * DIM + c]);
        ab[r][c] = f2bf(v);
    }
    __syncthreads();

    const int wave = __builtin_amdgcn_readfirstlane(tid) >> 5;
    const int lane = tid & 31;
    v16bf afr[4];
#pragma unroll
    for (int kt = 0; kt < 4; ++kt)
        afr[kt] = load_frag(&ab[0][0], DIM, 0, kt * 32, lane);

    for (int nt = wave; nt < 8; nt += 4) {
        v8f acc = {};
#pragma unroll
        for (int kt = 0; kt < 4; ++kt) {
            v16bf b = load_frag(wout_bf, DIM, nt * 16, kt * 32, lane);
            acc = wmma_bf16(afr[kt], b, acc);
        }
        const int n  = lane & 15;
        const int mh = (lane >> 4) * 8;
#pragma unroll
        for (int v = 0; v < 8; ++v)
            out[(size_t)(r0 + v + mh) * DIM + nt * 16 + n] = acc[v];
    }
}

// ---------------------------------------------------------------------------
// Host launcher
// inputs: 0:x 1:mask 2:ln1_w 3:ln1_b 4:Wl 5:Wr 6:Wlg 7:Wrg 8:Wog
//         9:ln2_w 10:ln2_b 11:Wout
// ---------------------------------------------------------------------------
extern "C" void kernel_launch(void* const* d_in, const int* in_sizes, int n_in,
                              void* d_out, int out_size, void* d_ws, size_t ws_size,
                              hipStream_t stream) {
    const float* x    = (const float*)d_in[0];
    const float* mask = (const float*)d_in[1];
    const float* l1w  = (const float*)d_in[2];
    const float* l1b  = (const float*)d_in[3];
    const float* Wl   = (const float*)d_in[4];
    const float* Wr   = (const float*)d_in[5];
    const float* Wlg  = (const float*)d_in[6];
    const float* Wrg  = (const float*)d_in[7];
    const float* Wog  = (const float*)d_in[8];
    const float* l2w  = (const float*)d_in[9];
    const float* l2b  = (const float*)d_in[10];
    const float* Wout = (const float*)d_in[11];
    float* out = (float*)d_out;

    char* ws = (char*)d_ws;
    unsigned short* wb      = (unsigned short*)(ws);                           // 192 KB
    unsigned short* left_t  = (unsigned short*)(ws + (size_t)1  * (1u << 20)); // 36 MB
    unsigned short* right_t = (unsigned short*)(ws + (size_t)40 * (1u << 20)); // 36 MB
    unsigned short* og      = (unsigned short*)(ws + (size_t)80 * (1u << 20)); // 36 MB
    float*          out_t   = (float*)        (ws + (size_t)122 * (1u << 20)); // 72 MB

    k0_convert_weights<<<dim3(98304 / 256), dim3(256), 0, stream>>>(
        Wl, Wr, Wlg, Wrg, Wog, Wout, wb);

    k1_ln_proj<<<dim3(NN / 16), dim3(128), 0, stream>>>(
        x, mask, l1w, l1b, wb, left_t, right_t, og);

    k2_einsum<<<dim3(N / 64, N / 64, DIM), dim3(128), 0, stream>>>(
        left_t, right_t, out_t);

    k3_out<<<dim3(NN / 16), dim3(128), 0, stream>>>(
        out_t, og, l2w, l2b, wb + 5 * 16384, out);
}
